// Encoder_41386304864955
// MI455X (gfx1250) — matrix operand order repaired and checked by reading, not compiled, not run on hardware
//
#include <hip/hip_runtime.h>
#include <stdint.h>

#define LSEQ  512
#define BATCH 128
#define EDIM  256
#define HDIM  128
#define GDIM  512   /* 4*HDIM */
#define VOCAB 32000

typedef __attribute__((ext_vector_type(16))) __bf16 v16bf;
typedef __attribute__((ext_vector_type(8)))  __bf16 v8bf;
typedef __attribute__((ext_vector_type(8)))  float  v8f;

// native f32 -> bf16 convert (gfx1250 has hardware bf16 cvt)
static __device__ __forceinline__ unsigned short f2bf_us(float f) {
    return __builtin_bit_cast(unsigned short, (__bf16)f);
}
static __device__ __forceinline__ v8bf ld8bf(const unsigned short* p) {
    uint4 u = *(const uint4*)p;                    // 8 bf16 = 16B
    return __builtin_bit_cast(v8bf, u);
}
static __device__ __forceinline__ v16bf cat16(v8bf lo, v8bf hi) {
    return __builtin_shufflevector(lo, hi, 0,1,2,3,4,5,6,7,8,9,10,11,12,13,14,15);
}
static __device__ __forceinline__ float sigf(float x) {
    return 1.0f / (1.0f + __expf(-x));
}

// ---------------------------------------------------------------------------
// f32 -> bf16 weight conversion (one-time prep)
// ---------------------------------------------------------------------------
__global__ void cvt_bf16(const float* __restrict__ src,
                         unsigned short* __restrict__ dst, int n) {
    int i = blockIdx.x * blockDim.x + threadIdx.x;
    if (i < n) dst[i] = f2bf_us(src[i]);
}

// ---------------------------------------------------------------------------
// Embedding gather -> contiguous bf16 x[L*B, E]. Each row is consumed by 16
// wave-tiles in the GEMM, so convert once here. One thread = 4 elements.
// ---------------------------------------------------------------------------
__global__ __launch_bounds__(256) void gather_x(
    const int* __restrict__ tokens, const float* __restrict__ emb,
    unsigned short* __restrict__ xbf)
{
    const int idx = blockIdx.x * blockDim.x + threadIdx.x;   // per 4 elems
    const int m   = idx >> 6;                                // EDIM/4 = 64 thr/row
    const int e4  = (idx & 63) << 2;
    const int tok = tokens[m];
    const unsigned q = (tok < 0) ? (unsigned)VOCAB : (unsigned)tok;
    float4 v = *(const float4*)(emb + (size_t)q * EDIM + e4);
    ushort4 o;
    o.x = f2bf_us(v.x); o.y = f2bf_us(v.y);
    o.z = f2bf_us(v.z); o.w = f2bf_us(v.w);
    *(ushort4*)(xbf + (size_t)m * EDIM + e4) = o;
}

// ---------------------------------------------------------------------------
// Input projection:  xg[m, n] = sum_e x[m, e] * W_ih[n, e] + bias[n]
// One wave = one 16(M) x 64(N) tile, K=256 fully unrolled, WMMA bf16.
// Wave id -> (mTile 0..4095, dir 0..1, nStrip 0..7)
// ---------------------------------------------------------------------------
__global__ __launch_bounds__(256) void xg_gemm(
    const unsigned short* __restrict__ xbf,
    const unsigned short* __restrict__ wih_f, const unsigned short* __restrict__ wih_b,
    const float* __restrict__ b_f, const float* __restrict__ b_b,
    float* __restrict__ xg_f, float* __restrict__ xg_b)
{
    const int lane   = threadIdx.x & 31;
    const int gw     = blockIdx.x * 8 + (threadIdx.x >> 5);
    const int mTile  = gw >> 4;
    const int rem    = gw & 15;
    const int dir    = rem >> 3;
    const int nStrip = rem & 7;
    const int m0     = mTile << 4;
    const int n0     = nStrip << 6;

    const unsigned short* wih  = dir ? wih_b : wih_f;
    const float*          bias = dir ? b_b   : b_f;
    float*                xg   = dir ? xg_b  : xg_f;

    const int ncol = lane & 15;     // A row / B col / D col within tile
    const int hi   = lane >> 4;     // lane half selects K sub-group & M half
    const int kA   = hi * 8;        // unused; A also loads contiguous K {kB .. kB+15}
    const int kB   = hi * 16;       // B: contiguous K {kB .. kB+15}

    const unsigned short* arow = xbf + (size_t)(m0 + ncol) * EDIM;

    v8f acc[4];
    #pragma unroll
    for (int j = 0; j < 4; ++j) {
        const float bv = bias[n0 + j * 16 + ncol];
        #pragma unroll
        for (int r = 0; r < 8; ++r) acc[j][r] = bv;
    }

    #pragma unroll
    for (int k0 = 0; k0 < EDIM; k0 += 32) {
        v16bf a = cat16(ld8bf(arow + k0 + kB), ld8bf(arow + k0 + kB + 8));
        #pragma unroll
        for (int j = 0; j < 4; ++j) {
            const unsigned short* wr =
                wih + (size_t)(n0 + j * 16 + ncol) * EDIM + k0 + kB;
            v16bf b = cat16(ld8bf(wr), ld8bf(wr + 8));
            acc[j] = __builtin_amdgcn_wmma_f32_16x16x32_bf16(
                false, a, false, b, (short)0, acc[j], false, false);
        }
    }

    #pragma unroll
    for (int j = 0; j < 4; ++j) {
        const int n = n0 + j * 16 + ncol;
        #pragma unroll
        for (int r = 0; r < 8; ++r)
            xg[(size_t)(m0 + r + hi * 8) * GDIM + n] = acc[j][r];
    }
}

// ---------------------------------------------------------------------------
// Recurrent scan. 16 blocks = 2 dirs x 8 batch-tiles of 16. 8 waves/block.
// Each wave pins its 64-column W_hh strip in 128 VGPRs for all 512 steps.
// h lives in LDS as bf16 (A fragment source); c stays in thread registers.
// acc doubles as the software-pipelined xg C-tile: it is reloaded with
// xg[t+1] immediately after the gates are published (acc is dead then), so
// the global-load latency hides behind the transcendental elementwise phase.
// ---------------------------------------------------------------------------
__global__ __launch_bounds__(256) void lstm_scan(
    const float* __restrict__ mask,
    const unsigned short* __restrict__ whh_f, const unsigned short* __restrict__ whh_b,
    const float* __restrict__ xg_f, const float* __restrict__ xg_b,
    float* __restrict__ out)
{
    __shared__ __align__(16) unsigned short hbf_s[16 * HDIM];   // 4 KB
    __shared__ __align__(16) float          gates_s[16 * GDIM]; // 32 KB

    const int dir = blockIdx.x >> 3;
    const int b0  = (blockIdx.x & 7) << 4;
    const unsigned short* whh = dir ? whh_b : whh_f;
    const float*          xg  = dir ? xg_b  : xg_f;

    const int tid  = threadIdx.x;
    const int lane = tid & 31;
    const int w    = tid >> 5;
    const int ncol = lane & 15;
    const int hi   = lane >> 4;

    // --- pin W_hh strip in registers: 4 N-tiles x 4 K-chunks -------------
    v16bf Bf[4][4];
    #pragma unroll
    for (int j = 0; j < 4; ++j) {
        const unsigned short* wr =
            whh + (size_t)(w * 64 + j * 16 + ncol) * HDIM + hi * 16;
        #pragma unroll
        for (int k = 0; k < 4; ++k)
            Bf[k][j] = cat16(ld8bf(wr + k * 32), ld8bf(wr + k * 32 + 8));
    }

    // --- init state ------------------------------------------------------
    float c_r[8];
    #pragma unroll
    for (int q = 0; q < 8; ++q) c_r[q] = 0.0f;
    #pragma unroll
    for (int q = 0; q < 8; ++q) hbf_s[tid * 8 + q] = 0;   // h = 0 (bf16)
    __syncthreads();

    const int em = tid >> 4;          // elementwise: row within batch tile
    const int en = (tid & 15) * 8;    // elementwise: col base (8 cells)

    // --- preload first timestep's xg C-tile ------------------------------
    v8f acc[4];
    {
        const int t0 = dir ? (LSEQ - 1) : 0;
        #pragma unroll
        for (int j = 0; j < 4; ++j) {
            const int n = w * 64 + j * 16 + ncol;
            const float* xr = xg + ((size_t)t0 * BATCH + b0 + hi * 8) * GDIM + n;
            #pragma unroll
            for (int r = 0; r < 8; ++r) acc[j][r] = xr[(size_t)r * GDIM];
        }
    }

    for (int step = 0; step < LSEQ; ++step) {
        const int t = dir ? (LSEQ - 1 - step) : step;

        // gates = h @ Whh^T + xg[t] : A from LDS bf16 h, B pinned in regs
        #pragma unroll
        for (int k = 0; k < 4; ++k) {
            const unsigned short* hr = hbf_s + ncol * HDIM + k * 32 + hi * 16;
            v16bf a = cat16(ld8bf(hr), ld8bf(hr + 8));
            #pragma unroll
            for (int j = 0; j < 4; ++j)
                acc[j] = __builtin_amdgcn_wmma_f32_16x16x32_bf16(
                    false, a, false, Bf[k][j], (short)0, acc[j], false, false);
        }

        // publish gates
        #pragma unroll
        for (int j = 0; j < 4; ++j) {
            const int n = w * 64 + j * 16 + ncol;
            #pragma unroll
            for (int r = 0; r < 8; ++r)
                gates_s[(r + hi * 8) * GDIM + n] = acc[j][r];
        }
        __syncthreads();

        // acc is dead now: refill it with the next timestep's xg C-tile.
        // These loads retire while the elementwise phase below executes.
        const int tn = dir ? (t > 0 ? t - 1 : 0)
                           : (t < LSEQ - 1 ? t + 1 : LSEQ - 1);
        #pragma unroll
        for (int j = 0; j < 4; ++j) {
            const int n = w * 64 + j * 16 + ncol;
            const float* xr = xg + ((size_t)tn * BATCH + b0 + hi * 8) * GDIM + n;
            #pragma unroll
            for (int r = 0; r < 8; ++r) acc[j][r] = xr[(size_t)r * GDIM];
        }

        // elementwise LSTM cell: 8 cells per thread
        const float mval = mask[(size_t)t * BATCH + b0 + em];
        float* orow = out + ((size_t)t * BATCH + b0 + em) * (2 * HDIM)
                          + dir * HDIM + en;
        #pragma unroll
        for (int q = 0; q < 8; ++q) {
            const int n = en + q;
            const float gi = gates_s[em * GDIM + n];
            const float gf = gates_s[em * GDIM + HDIM     + n];
            const float gg = gates_s[em * GDIM + 2 * HDIM + n];
            const float go = gates_s[em * GDIM + 3 * HDIM + n];
            float c = sigf(gf) * c_r[q] + sigf(gi) * tanhf(gg);
            float h = sigf(go) * tanhf(c);
            h *= mval; c *= mval;
            c_r[q] = c;
            orow[q] = h;
            hbf_s[em * HDIM + n] = f2bf_us(h);
        }
        __syncthreads();
    }
}

// ---------------------------------------------------------------------------
extern "C" void kernel_launch(void* const* d_in, const int* in_sizes, int n_in,
                              void* d_out, int out_size, void* d_ws, size_t ws_size,
                              hipStream_t stream) {
    (void)in_sizes; (void)n_in; (void)out_size; (void)ws_size;
    const int*   tokens = (const int*)  d_in[0];
    const float* mask   = (const float*)d_in[1];
    const float* emb    = (const float*)d_in[2];
    const float* W_ih_f = (const float*)d_in[3];
    const float* W_hh_f = (const float*)d_in[4];
    const float* b_f    = (const float*)d_in[5];
    const float* W_ih_b = (const float*)d_in[6];
    const float* W_hh_b = (const float*)d_in[7];
    const float* b_b    = (const float*)d_in[8];
    float* out = (float*)d_out;

    // workspace: xg_f | xg_b (fp32) | xbf | bf16 weight copies
    const size_t xgElems = (size_t)LSEQ * BATCH * GDIM;
    float* xg_f = (float*)d_ws;
    float* xg_b = xg_f + xgElems;
    unsigned short* xbf   = (unsigned short*)(xg_b + xgElems);
    unsigned short* wih_f = xbf + (size_t)LSEQ * BATCH * EDIM;
    unsigned short* wih_b = wih_f + (size_t)GDIM * EDIM;
    unsigned short* whh_f = wih_b + (size_t)GDIM * EDIM;
    unsigned short* whh_b = whh_f + (size_t)GDIM * HDIM;

    cvt_bf16<<<(GDIM * EDIM + 255) / 256, 256, 0, stream>>>(W_ih_f, wih_f, GDIM * EDIM);
    cvt_bf16<<<(GDIM * EDIM + 255) / 256, 256, 0, stream>>>(W_ih_b, wih_b, GDIM * EDIM);
    cvt_bf16<<<(GDIM * HDIM + 255) / 256, 256, 0, stream>>>(W_hh_f, whh_f, GDIM * HDIM);
    cvt_bf16<<<(GDIM * HDIM + 255) / 256, 256, 0, stream>>>(W_hh_b, whh_b, GDIM * HDIM);

    // embedding gather + bf16 convert (once per row, not once per wave-tile)
    gather_x<<<(LSEQ * BATCH * (EDIM / 4)) / 256, 256, 0, stream>>>(tokens, emb, xbf);

    // 65536 wave-tiles (4096 mTiles x 2 dirs x 8 nStrips), 8 waves/block
    xg_gemm<<<8192, 256, 0, stream>>>(xbf, wih_f, wih_b, b_f, b_b, xg_f, xg_b);

    // 2 dirs x 8 batch-tiles
    lstm_scan<<<16, 256, 0, stream>>>(mask, whh_f, whh_b, xg_f, xg_b, out);
}
